// GATLayerAdjMatrix_23682449670190
// MI455X (gfx1250) — matrix operand order repaired and checked by reading, not compiled, not run on hardware
//
#include <hip/hip_runtime.h>
#include <hip/hip_bf16.h>

typedef __attribute__((ext_vector_type(16))) __bf16 bf16x16;
typedef __attribute__((ext_vector_type(8)))  __bf16 bf16x8;
typedef __attribute__((ext_vector_type(8)))  float  f32x8;
typedef __attribute__((ext_vector_type(4)))  float  f32x4;
typedef __attribute__((ext_vector_type(4)))  int    i32x4;

#define GAT_ALPHA 0.2f
#define GAT_NEGINF (-1.0e10f)
#define NN   4096      // N nodes
#define FIN  512
#define NH   8
#define FOUT 64
#define HF   (NH*FOUT) // 512
#define BSTR 40        // padded LDS row stride (bf16): 80B = 20 banks, 16B aligned

union ABf16 { bf16x16 v; bf16x8 h[2]; __bf16 e[16]; };

// --------------------------------------------------------------------------
// Async global->LDS staging (CDNA5 path) with portable fallback.
#if defined(__has_builtin)
#if __has_builtin(__builtin_amdgcn_global_load_async_to_lds_b128)
#define HAVE_ASYNC_LDS 1
#endif
#endif

#ifdef HAVE_ASYNC_LDS
typedef __attribute__((address_space(1))) i32x4 as1_i32x4;
typedef __attribute__((address_space(3))) i32x4 as3_i32x4;
__device__ __forceinline__ void stage_b128(const __bf16* g, __bf16* l) {
    as1_i32x4* gp = (as1_i32x4*)(i32x4*)const_cast<__bf16*>(g);
    as3_i32x4* lp = (as3_i32x4*)(i32x4*)l;
    __builtin_amdgcn_global_load_async_to_lds_b128(gp, lp, 0, 0);
}
__device__ __forceinline__ void wait_async_le2() {
    asm volatile("s_wait_asynccnt 2" ::: "memory");
}
__device__ __forceinline__ void wait_async_0() {
    asm volatile("s_wait_asynccnt 0" ::: "memory");
}
#else
__device__ __forceinline__ void stage_b128(const __bf16* g, __bf16* l) {
    *(bf16x8*)l = *(const bf16x8*)g;   // load + ds_store fallback
}
__device__ __forceinline__ void wait_async_le2() {}
__device__ __forceinline__ void wait_async_0()   {}
#endif

// ---------------------------------------------------------------------------
// 0a. x (f32) -> xbf (bf16), row-major [NN][FIN]
__global__ __launch_bounds__(256) void cvt_x(const float* __restrict__ x,
                                             __bf16* __restrict__ xbf, int n) {
    int i = blockIdx.x * 256 + threadIdx.x;
    if (i < n) xbf[i] = (__bf16)x[i];
}

// 0b. W (f32, [FIN][HF]) -> WT bf16 [HF][FIN]  (column-major for B-operand loads)
__global__ __launch_bounds__(256) void cvt_wT(const float* __restrict__ W,
                                              __bf16* __restrict__ wtbf) {
    int idx = blockIdx.x * 256 + threadIdx.x;   // FIN*HF
    int r = idx >> 9;
    int c = idx & 511;
    wtbf[(size_t)c * FIN + r] = (__bf16)W[idx];
}

__global__ __launch_bounds__(256) void zero_out(float* __restrict__ out, int n) {
    int i = blockIdx.x * 256 + threadIdx.x;
    if (i < n) out[i] = 0.0f;
}

// ---------------------------------------------------------------------------
// 1. Wh = x @ W, written as WhT bf16 [col = h*64+f][row = node]
__global__ __launch_bounds__(32) void gemm_wh(const __bf16* __restrict__ xbf,
                                              const __bf16* __restrict__ wtbf,
                                              __bf16* __restrict__ whT) {
    const int lane = threadIdx.x;
    const int hf   = lane >> 4;
    const int l16  = lane & 15;
    const int i0   = blockIdx.x * 16;
    const int n0   = blockIdx.y * 16;

    const __bf16* arow = xbf  + (size_t)(i0 + l16) * FIN;
    const __bf16* brow = wtbf + (size_t)(n0 + l16) * FIN;

    f32x8 acc = {};
    for (int k0 = 0; k0 < FIN; k0 += 32) {
        ABf16 A;
        A.h[0] = *(const bf16x8*)(arow + k0 + hf * 8);
        A.h[1] = *(const bf16x8*)(arow + k0 + hf * 8 + 16);
        ABf16 B;
        B.h[0] = *(const bf16x8*)(brow + k0 + hf * 8);
        B.h[1] = *(const bf16x8*)(brow + k0 + hf * 8 + 16);
        acc = __builtin_amdgcn_wmma_f32_16x16x32_bf16(false, A.v, false, B.v,
                                                      (short)0, acc, false, false);
    }
    const size_t col = (size_t)(n0 + l16);
#pragma unroll
    for (int r = 0; r < 8; ++r) {
        int row = i0 + r + hf * 8;
        whT[col * NN + row] = (__bf16)acc[r];
    }
}

// ---------------------------------------------------------------------------
// 2. f1t[h][n], f2t[h][n] (head-major)
__global__ __launch_bounds__(256) void gat_f12(const __bf16* __restrict__ whT,
                                               const float* __restrict__ a,
                                               float* __restrict__ f1t,
                                               float* __restrict__ f2t) {
    int idx = blockIdx.x * 256 + threadIdx.x;   // NN*NH
    int h = idx >> 12;
    int n = idx & (NN - 1);
    const __bf16* base = whT + (size_t)h * FOUT * NN + n;
    float s1 = 0.f, s2 = 0.f;
#pragma unroll 8
    for (int f = 0; f < FOUT; ++f) {
        float w = (float)base[(size_t)f * NN];
        s1 += w * a[f];
        s2 += w * a[FOUT + f];
    }
    f1t[idx] = s1;
    f2t[idx] = s2;
}

// ---------------------------------------------------------------------------
// 3. Softmax stats.  Monotonicity of leaky_relu:
//    max_j e[i,j,h] = lrelu(f1 + max masked f2[j])  (or NEG_INF if row empty),
//    so pass 1 is a 2-op/elem masked max (no exps) and pass 2 is 1 exp/elem.
//    Block = row i; wave w = head w (adj row shared through WGP$).
__global__ __launch_bounds__(256) void gat_stats(const int* __restrict__ adj,
                                                 const float* __restrict__ f1t,
                                                 const float* __restrict__ f2t,
                                                 float* __restrict__ mrow,
                                                 float* __restrict__ lrow) {
    const int i    = blockIdx.x;
    const int h    = threadIdx.x >> 5;
    const int lane = threadIdx.x & 31;

    const float f1 = f1t[h * NN + i];
    const int*   ap = adj + (size_t)i * NN;
    const float* fp = f2t + (size_t)h * NN;

    // pass 1: masked max of f2 over the row
    float mf = -3.0e38f;
    for (int j4 = lane * 4; j4 < NN; j4 += 128) {
        i32x4 av = *(const i32x4*)(ap + j4);
        f32x4 fv = *(const f32x4*)(fp + j4);
#pragma unroll
        for (int k = 0; k < 4; ++k)
            mf = fmaxf(mf, av[k] > 0 ? fv[k] : -3.0e38f);
    }
#pragma unroll
    for (int off = 16; off > 0; off >>= 1)
        mf = fmaxf(mf, __shfl_xor(mf, off, 32));
    float mm = f1 + mf;
    mm = mm > 0.f ? mm : GAT_ALPHA * mm;
    const float m = (mf > -1.0e38f) ? mm : GAT_NEGINF;

    // pass 2: sum of exp(e - m)
    float l = 0.f;
    for (int j4 = lane * 4; j4 < NN; j4 += 128) {
        i32x4 av = *(const i32x4*)(ap + j4);
        f32x4 fv = *(const f32x4*)(fp + j4);
#pragma unroll
        for (int k = 0; k < 4; ++k) {
            float e = f1 + fv[k];
            e = e > 0.f ? e : GAT_ALPHA * e;
            e = av[k] > 0 ? e : GAT_NEGINF;
            l += __expf(e - m);
        }
    }
#pragma unroll
    for (int off = 16; off > 0; off >>= 1)
        l += __shfl_xor(l, off, 32);
    if (lane == 0) { mrow[h * NN + i] = m; lrow[h * NN + i] = l; }
}

// ---------------------------------------------------------------------------
// 4. Fused softmax-weights + aggregation.  Block = 4 waves sharing (h, j-chunk);
//    wave w handles i-tile blockIdx.x*64 + w*16.  The 32x64 bf16 B tile is
//    staged once per block into double-buffered LDS via async global->LDS DMA
//    and consumed by all 4 waves (4x less L2 B traffic; DMA overlaps the
//    VALU-bound exp/p computation).
__device__ __forceinline__ void pvals4(i32x4 av, f32x4 gv, float f1, float m,
                                       float scale, __bf16* dst) {
#pragma unroll
    for (int k = 0; k < 4; ++k) {
        float e = f1 + gv[k];
        e = e > 0.f ? e : GAT_ALPHA * e;
        e = av[k] > 0 ? e : GAT_NEGINF;
        dst[k] = (__bf16)(__expf(e - m) * scale);
    }
}

__global__ __launch_bounds__(128) void gat_agg(const int* __restrict__ adj,
                                               const __bf16* __restrict__ whT,
                                               const float* __restrict__ f1t,
                                               const float* __restrict__ f2t,
                                               const float* __restrict__ mrow,
                                               const float* __restrict__ lrow,
                                               float* __restrict__ out) {
    __shared__ __bf16 bs[2][64 * BSTR];          // 2 x 5.0 KB double buffer

    const int tid  = threadIdx.x;
    const int wave = tid >> 5;
    const int lane = tid & 31;
    const int hf   = lane >> 4;
    const int l16  = lane & 15;
    const int i0   = blockIdx.x * 64 + wave * 16;
    const int h    = blockIdx.y;
    const int jc   = blockIdx.z;                 // 4 chunks of 1024 columns
    const int jbase = jc * 1024;

    // staging assignment: thread t moves 32B of B tile: f-row = t>>1, j-part = (t&1)*16
    const int bf_row = tid >> 1;                 // 0..63
    const int bj0    = (tid & 1) * 16;           // 0 | 16
    const __bf16* gB = whT + (size_t)(h * FOUT + bf_row) * NN + bj0;
    __bf16* lB[2] = { &bs[0][bf_row * BSTR + bj0], &bs[1][bf_row * BSTR + bj0] };

    const int row  = i0 + l16;                   // A-operand row for this lane
    const float f1 = f1t[h * NN + row];
    const float m  = mrow[h * NN + row];
    const float sc = 1.0f / (8.0f * lrow[h * NN + row]);

    const int kb1 = hf * 8;                      // K  0..7  | 8..15
    const int kb2 = 16 + hf * 8;                 // K 16..23 | 24..31

    const int*   arow = adj + (size_t)row * NN;
    const float* f2p  = f2t + (size_t)h * NN;

    f32x8 acc0 = {}, acc1 = {}, acc2 = {}, acc3 = {};

    // prologue: stage tile 0 into buffer 0
    stage_b128(gB + jbase,     lB[0]);
    stage_b128(gB + jbase + 8, lB[0] + 8);

    for (int it = 0; it < 32; ++it) {
        const int j0 = jbase + it * 32;

        // stage tile it+1 into the other buffer (overlaps with p computation)
        if (it < 31) {
            stage_b128(gB + j0 + 32,     lB[(it + 1) & 1]);
            stage_b128(gB + j0 + 32 + 8, lB[(it + 1) & 1] + 8);
        }

        __builtin_prefetch(arow + j0 + 32, 0, 1);

        i32x4 a0 = *(const i32x4*)(arow + j0 + kb1);
        i32x4 a1 = *(const i32x4*)(arow + j0 + kb1 + 4);
        i32x4 a2 = *(const i32x4*)(arow + j0 + kb2);
        i32x4 a3 = *(const i32x4*)(arow + j0 + kb2 + 4);
        f32x4 g0 = *(const f32x4*)(f2p + j0 + kb1);
        f32x4 g1 = *(const f32x4*)(f2p + j0 + kb1 + 4);
        f32x4 g2 = *(const f32x4*)(f2p + j0 + kb2);
        f32x4 g3 = *(const f32x4*)(f2p + j0 + kb2 + 4);

        ABf16 A;
        pvals4(a0, g0, f1, m, sc, A.e + 0);
        pvals4(a1, g1, f1, m, sc, A.e + 4);
        pvals4(a2, g2, f1, m, sc, A.e + 8);
        pvals4(a3, g3, f1, m, sc, A.e + 12);

        // tile it's DMA (issued last iteration) is complete once asynccnt <= 2
        if (it < 31) wait_async_le2(); else wait_async_0();
        __syncthreads();

        // B operand from LDS: f row = t*16+l16, K = hf*8+{0..7} | 16+hf*8+{0..7} (b128 x2)
        const __bf16* lb = &bs[it & 1][(size_t)l16 * BSTR + hf * 8];
        ABf16 B0, B1, B2, B3;
        B0.h[0] = *(const bf16x8*)(lb);                B0.h[1] = *(const bf16x8*)(lb + 16);
        B1.h[0] = *(const bf16x8*)(lb + 16 * BSTR);    B1.h[1] = *(const bf16x8*)(lb + 16 * BSTR + 16);
        B2.h[0] = *(const bf16x8*)(lb + 32 * BSTR);    B2.h[1] = *(const bf16x8*)(lb + 32 * BSTR + 16);
        B3.h[0] = *(const bf16x8*)(lb + 48 * BSTR);    B3.h[1] = *(const bf16x8*)(lb + 48 * BSTR + 16);

        acc0 = __builtin_amdgcn_wmma_f32_16x16x32_bf16(false, A.v, false, B0.v, (short)0, acc0, false, false);
        acc1 = __builtin_amdgcn_wmma_f32_16x16x32_bf16(false, A.v, false, B1.v, (short)0, acc1, false, false);
        acc2 = __builtin_amdgcn_wmma_f32_16x16x32_bf16(false, A.v, false, B2.v, (short)0, acc2, false, false);
        acc3 = __builtin_amdgcn_wmma_f32_16x16x32_bf16(false, A.v, false, B3.v, (short)0, acc3, false, false);

        __syncthreads();   // protect buffer it&1 before next overwrite
    }

#pragma unroll
    for (int r = 0; r < 8; ++r) {
        float* orow = out + (size_t)(i0 + r + hf * 8) * FOUT + l16;
        atomicAdd(orow +  0, acc0[r]);
        atomicAdd(orow + 16, acc1[r]);
        atomicAdd(orow + 32, acc2[r]);
        atomicAdd(orow + 48, acc3[r]);
    }
}

// ---------------------------------------------------------------------------
extern "C" void kernel_launch(void* const* d_in, const int* in_sizes, int n_in,
                              void* d_out, int out_size, void* d_ws, size_t ws_size,
                              hipStream_t stream) {
    const float* x   = (const float*)d_in[0];   // [4096][512]
    const int*   adj = (const int*)  d_in[1];   // [4096][4096]
    const float* W   = (const float*)d_in[2];   // [512][512]
    const float* a   = (const float*)d_in[3];   // [128]
    float* out = (float*)d_out;                 // [4096][64]

    char* ws = (char*)d_ws;
    __bf16* xbf  = (__bf16*)(ws);                             // 4 MB
    __bf16* wtbf = (__bf16*)(ws + (4u << 20));                // 512 KB
    __bf16* whT  = (__bf16*)(ws + (4u << 20) + (512u << 10)); // 4 MB
    char* fbase  = ws + (8u << 20) + (512u << 10);
    float* f1t  = (float*)(fbase);
    float* f2t  = (float*)(fbase + 131072);
    float* mrow = (float*)(fbase + 2 * 131072);
    float* lrow = (float*)(fbase + 3 * 131072);

    cvt_x   <<<dim3((NN * FIN) / 256), dim3(256), 0, stream>>>(x, xbf, NN * FIN);
    cvt_wT  <<<dim3((FIN * HF) / 256), dim3(256), 0, stream>>>(W, wtbf);
    zero_out<<<dim3((NN * FOUT) / 256), dim3(256), 0, stream>>>(out, NN * FOUT);

    gemm_wh  <<<dim3(NN / 16, HF / 16), dim3(32), 0, stream>>>(xbf, wtbf, whT);
    gat_f12  <<<dim3((NN * NH) / 256), dim3(256), 0, stream>>>(whT, a, f1t, f2t);
    gat_stats<<<dim3(NN), dim3(256), 0, stream>>>(adj, f1t, f2t, mrow, lrow);
    gat_agg  <<<dim3(NN / 64, NH, 4), dim3(128), 0, stream>>>(adj, whT, f1t, f2t,
                                                              mrow, lrow, out);
}
